// ConcatAttention_70746701300158
// MI455X (gfx1250) — compile-verified, matrix operand order checked by reading, not run on hardware
//
#include <hip/hip_runtime.h>
#include <hip/hip_bf16.h>

// Problem sizes (fixed by the reference).
#define Bsz 32
#define Ssz 2048
#define Hsz 1024

// Stage-1 blocking.
#define MROWS 64          // keys rows per workgroup
#define ROWDW 520         // LDS row stride in dwords (1040 bf16); 520%64==8 -> b128 tiles banks 2x, 16B aligned
#define KSTEP 32          // bf16 WMMA K

typedef __attribute__((ext_vector_type(16))) __bf16    v16bf;
typedef __attribute__((ext_vector_type(8)))  float     v8f;
typedef __attribute__((ext_vector_type(4)))  unsigned  u32x4;

union ABfrag { u32x4 q[2]; unsigned u[8]; v16bf v; };

// ---------- helpers ----------
__device__ __forceinline__ unsigned bf16rne(float x) {
    unsigned u = __float_as_uint(x);
    u += 0x7fffu + ((u >> 16) & 1u);        // round-to-nearest-even
    return u >> 16;
}
__device__ __forceinline__ unsigned pack2(float a, float b) {
    return bf16rne(a) | (bf16rne(b) << 16);
}
__device__ __forceinline__ float fast_tanh(float x) {
    // tanh(x) = 1 - 2/(exp(2x)+1), exp via single v_exp_f32, div via v_rcp_f32
    float cx = fminf(fmaxf(x, -10.f), 10.f);
    float e  = __builtin_amdgcn_exp2f(cx * 2.885390081777927f);  // exp(2x)=2^(2x*log2 e)
    return 1.f - 2.f * __builtin_amdgcn_rcpf(e + 1.f);
}

// ---------- stage 0a: qW2[b,d] = query[b,:] @ W2[:,d] ----------
__global__ __launch_bounds__(256) void qw2_kernel(const float* __restrict__ query,
                                                  const float* __restrict__ W2,
                                                  float* __restrict__ qw2) {
    int b = blockIdx.x;
    int d = blockIdx.y * 256 + threadIdx.x;
    const float* q = query + (size_t)b * Hsz;
    float acc = 0.f;
    for (int h = 0; h < Hsz; ++h) acc += q[h] * W2[(size_t)h * Hsz + d];   // q[h] scalarizes
    qw2[(size_t)b * Hsz + d] = acc;
}

// ---------- stage 0b: W1t[d][k] = bf16(W1[k][d]) (transpose + convert, LDS tiled) ----------
__global__ __launch_bounds__(256) void w1t_kernel(const float* __restrict__ W1,
                                                  unsigned short* __restrict__ W1t) {
    __shared__ float tile[32][33];
    int n0 = blockIdx.x * 32, k0 = blockIdx.y * 32;
    int tx = threadIdx.x, ty = threadIdx.y;     // block (32,8)
    for (int j = 0; j < 32; j += 8)
        tile[ty + j][tx] = W1[(size_t)(k0 + ty + j) * Hsz + n0 + tx];
    __syncthreads();
    for (int j = 0; j < 32; j += 8) {
        int n = n0 + ty + j, k = k0 + tx;
        W1t[(size_t)n * Hsz + k] = (unsigned short)bf16rne(tile[tx][ty + j]);
    }
}

// ---------- stage 1: fused keys@W1 (+qW2, tanh, *V, row-reduce, mask) -> scores ----------
__global__ __launch_bounds__(256) void scores_kernel(const float* __restrict__ keys,
                                                     const unsigned short* __restrict__ W1t,
                                                     const float* __restrict__ qw2,
                                                     const float* __restrict__ Vvec,
                                                     const int* __restrict__ mask,
                                                     float* __restrict__ scores) {
    extern __shared__ unsigned smem[];                  // [MROWS*ROWDW] bf16 A-tile + [8*64] f32 partials
    float* part = (float*)(smem + MROWS * ROWDW);

    const int tiles_per_b = Ssz / MROWS;                // 32
    const int b      = blockIdx.x / tiles_per_b;
    const int s0     = (blockIdx.x % tiles_per_b) * MROWS;
    const int tid    = threadIdx.x;
    const int wave   = tid >> 5;
    const int lane   = tid & 31;
    const int lane15 = lane & 15;
    const int kHalfD = (lane >> 4) << 2;                // hi half-wave: +8 bf16 = +4 dwords

    // --- load keys tile [64 x 1024] f32 -> bf16 LDS (coalesced float4, b64 LDS stores) ---
    const float* kb = keys + ((size_t)(b * Ssz + s0)) * Hsz;
    for (int i = tid; i < MROWS * (Hsz / 4); i += 256) {
        int row = i >> 8;                               // Hsz/4 = 256 float4 per row
        int c4  = i & 255;
        float4 f = ((const float4*)(kb + (size_t)row * Hsz))[c4];
        unsigned* d = smem + row * ROWDW + c4 * 2;
        d[0] = pack2(f.x, f.y);
        d[1] = pack2(f.z, f.w);
    }
    __syncthreads();

    float rowsum[4][8];
#pragma unroll
    for (int t = 0; t < 4; ++t)
#pragma unroll
        for (int r = 0; r < 8; ++r) rowsum[t][r] = 0.f;

    // 4 column chunks: each chunk covers 256 cols = 8 waves * 2 strips * 16
    for (int chunk = 0; chunk < 4; ++chunk) {
        const int ct0  = chunk * 16 + wave * 2;
        const int col0 = ct0 * 16 + lane15;
        const int col1 = col0 + 16;
        const unsigned* w0p = (const unsigned*)(W1t + (size_t)col0 * Hsz);
        const unsigned* w1p = (const unsigned*)(W1t + (size_t)col1 * Hsz);

        v8f acc0[4] = {};
        v8f acc1[4] = {};

#pragma unroll 2
        for (int k = 0; k < Hsz; k += KSTEP) {
            const int kd = (k >> 1) + kHalfD;
            ABfrag B0, B1;
            B0.q[0] = *(const u32x4*)(w0p + kd);
            B0.q[1] = *(const u32x4*)(w0p + kd + 8);
            B1.q[0] = *(const u32x4*)(w1p + kd);
            B1.q[1] = *(const u32x4*)(w1p + kd + 8);
#pragma unroll
            for (int t = 0; t < 4; ++t) {
                ABfrag A;
                const unsigned* ar = smem + (t * 16 + lane15) * ROWDW + kd;
                A.q[0] = *(const u32x4*)ar;             // ds_load_b128
                A.q[1] = *(const u32x4*)(ar + 8);
                acc0[t] = __builtin_amdgcn_wmma_f32_16x16x32_bf16(
                    false, A.v, false, B0.v, (short)0, acc0[t], false, false);
                acc1[t] = __builtin_amdgcn_wmma_f32_16x16x32_bf16(
                    false, A.v, false, B1.v, (short)0, acc1[t], false, false);
            }
        }

        // epilogue for this column pair: scores += tanh(acc + qW2[col]) * V[col]
        const float q0 = qw2[b * Hsz + col0], q1 = qw2[b * Hsz + col1];
        const float v0 = Vvec[col0],          v1 = Vvec[col1];
#pragma unroll
        for (int t = 0; t < 4; ++t)
#pragma unroll
            for (int r = 0; r < 8; ++r) {
                rowsum[t][r] += fast_tanh(acc0[t][r] + q0) * v0
                              + fast_tanh(acc1[t][r] + q1) * v1;
            }
    }

    // deterministic reduction: shfl-xor across the 16 lanes sharing a row set,
    // then fixed-order sum across waves.
#pragma unroll
    for (int t = 0; t < 4; ++t)
#pragma unroll
        for (int r = 0; r < 8; ++r) {
            float v = rowsum[t][r];
            v += __shfl_xor(v, 1);
            v += __shfl_xor(v, 2);
            v += __shfl_xor(v, 4);
            v += __shfl_xor(v, 8);
            if (lane15 == 0) {
                int row = t * 16 + r + ((lane >> 4) << 3);   // C/D layout: hi half-wave = rows +8
                part[wave * 64 + row] = v;
            }
        }
    __syncthreads();

    if (tid < 64) {
        float v = 0.f;
        for (int w = 0; w < 8; ++w) v += part[w * 64 + tid];
        int s = s0 + tid;
        int m = mask[b * Ssz + s];
        scores[b * Ssz + s] = m ? v : -1e9f;
    }
}

// ---------- stage 2: softmax over S per batch ----------
__global__ __launch_bounds__(256) void softmax_kernel(const float* __restrict__ scores,
                                                      float* __restrict__ attn) {
    __shared__ float red[256];
    int b = blockIdx.x, tid = threadIdx.x;
    float x[8];
    float mx = -3.4e38f;
#pragma unroll
    for (int i = 0; i < 8; ++i) {
        x[i] = scores[b * Ssz + tid + i * 256];
        mx = fmaxf(mx, x[i]);
    }
    red[tid] = mx; __syncthreads();
    for (int o = 128; o > 0; o >>= 1) {
        if (tid < o) red[tid] = fmaxf(red[tid], red[tid + o]);
        __syncthreads();
    }
    mx = red[0]; __syncthreads();
    float sum = 0.f;
#pragma unroll
    for (int i = 0; i < 8; ++i) {
        x[i] = __builtin_amdgcn_exp2f((x[i] - mx) * 1.4426950408889634f);
        sum += x[i];
    }
    red[tid] = sum; __syncthreads();
    for (int o = 128; o > 0; o >>= 1) {
        if (tid < o) red[tid] += red[tid + o];
        __syncthreads();
    }
    float inv = 1.f / red[0];
#pragma unroll
    for (int i = 0; i < 8; ++i) attn[b * Ssz + tid + i * 256] = x[i] * inv;
}

// ---------- stage 3a: partial context over S chunks (deterministic, no atomics) ----------
__global__ __launch_bounds__(256) void ctx_partial_kernel(const float* __restrict__ keys,
                                                          const float* __restrict__ attn,
                                                          float* __restrict__ partial) {
    int b = blockIdx.x, chunk = blockIdx.y;          // 16 chunks of 128 s-values
    int h0 = threadIdx.x * 4;
    int sBase = chunk * (Ssz / 16);
    const float* kb = keys + ((size_t)b * Ssz + sBase) * Hsz;
    const float* ab = attn + (size_t)b * Ssz + sBase;
    float4 acc = {0.f, 0.f, 0.f, 0.f};
    for (int s = 0; s < Ssz / 16; ++s) {
        float a = ab[s];
        float4 kv = *(const float4*)(kb + (size_t)s * Hsz + h0);
        acc.x += a * kv.x; acc.y += a * kv.y; acc.z += a * kv.z; acc.w += a * kv.w;
    }
    *(float4*)(partial + ((size_t)(b * 16 + chunk)) * Hsz + h0) = acc;
}

// ---------- stage 3b: reduce partials -> context ----------
__global__ __launch_bounds__(256) void ctx_reduce_kernel(const float* __restrict__ partial,
                                                         float* __restrict__ ctx) {
    int b = blockIdx.x;
    int h0 = threadIdx.x * 4;
    float4 acc = {0.f, 0.f, 0.f, 0.f};
    for (int c = 0; c < 16; ++c) {
        float4 p = *(const float4*)(partial + ((size_t)(b * 16 + c)) * Hsz + h0);
        acc.x += p.x; acc.y += p.y; acc.z += p.z; acc.w += p.w;
    }
    *(float4*)(ctx + (size_t)b * Hsz + h0) = acc;
}

// ---------- launcher ----------
extern "C" void kernel_launch(void* const* d_in, const int* in_sizes, int n_in,
                              void* d_out, int out_size, void* d_ws, size_t ws_size,
                              hipStream_t stream) {
    (void)in_sizes; (void)n_in; (void)out_size; (void)ws_size;

    const float* query = (const float*)d_in[0];   // [B,1,H]
    const float* keys  = (const float*)d_in[1];   // [B,S,H]
    const int*   mask  = (const int*)d_in[2];     // [B,S,1]
    const float* W1    = (const float*)d_in[3];   // [H,H]
    const float* W2    = (const float*)d_in[4];   // [H,H]
    const float* Vv    = (const float*)d_in[5];   // [H,1]

    float* ctx_out  = (float*)d_out;              // [B,H]
    float* attn_out = ctx_out + Bsz * Hsz;        // [B,S]

    char* ws = (char*)d_ws;
    unsigned short* W1t = (unsigned short*)ws;                         // 2 MB bf16
    float* qw2     = (float*)(ws + (size_t)Hsz * Hsz * 2);             // 128 KB
    float* scores  = qw2 + Bsz * Hsz;                                  // 256 KB
    float* partial = scores + Bsz * Ssz;                               // 2 MB

    w1t_kernel<<<dim3(Hsz / 32, Hsz / 32), dim3(32, 8), 0, stream>>>(W1, W1t);
    qw2_kernel<<<dim3(Bsz, Hsz / 256), 256, 0, stream>>>(query, W2, qw2);

    const size_t smemBytes = (MROWS * ROWDW + 8 * 64) * sizeof(unsigned);   // ~135 KB of the 320 KB WGP LDS
    scores_kernel<<<dim3((Bsz * Ssz) / MROWS), 256, smemBytes, stream>>>(
        keys, W1t, qw2, Vv, mask, scores);

    softmax_kernel<<<dim3(Bsz), 256, 0, stream>>>(scores, attn_out);
    ctx_partial_kernel<<<dim3(Bsz, 16), 256, 0, stream>>>(keys, attn_out, partial);
    ctx_reduce_kernel<<<dim3(Bsz), 256, 0, stream>>>(partial, ctx_out);
}
